// mult_h_attention_82188494176269
// MI455X (gfx1250) — matrix operand order repaired and checked by reading, not compiled, not run on hardware
//
#include <hip/hip_runtime.h>
#include <hip/hip_bf16.h>

// ---------------------------------------------------------------------------
// Multi-head attention, CDNA5 (gfx1250) WMMA implementation.
//   EMBED=128, HEADS=8 (head_dim = 128), B=2, S=2048.
// All GEMMs run on v_wmma_f32_16x16x32_f16 (f16 in, f32 accumulate).
// Flash-attention K/V tiles are staged with GLOBAL_LOAD_ASYNC_TO_LDS_B128
// (ASYNCcnt-tracked, double-buffered).
// ---------------------------------------------------------------------------

typedef __attribute__((ext_vector_type(16))) _Float16 v16h;
typedef __attribute__((ext_vector_type(8)))  float    v8f;

union AFrag {
    v16h h;
    unsigned int u[8];
};

__device__ __forceinline__ v8f wmma16(const AFrag& a, const AFrag& b, v8f c) {
    return __builtin_amdgcn_wmma_f32_16x16x32_f16(
        /*neg_a=*/false, a.h, /*neg_b=*/false, b.h,
        /*c_mod=*/(short)0, c, /*reuse_a=*/false, /*reuse_b=*/false);
}

#define B_N   2
#define S_LEN 2048
#define HEADS 8
#define DIM   128
#define HDIM  1024   // HEADS*DIM

// ---------------------------------------------------------------------------
// Async global -> LDS copy of one 8192-byte tile (32 keys x 128 d, f16).
// 128 threads x 4 issues x 16B/lane = 8192B.  Tracked by ASYNCcnt.
// ---------------------------------------------------------------------------
__device__ __forceinline__ void stage_tile_async(const _Float16* gbase,
                                                 const _Float16* lbase,
                                                 int tid) {
    const unsigned long long g = (unsigned long long)gbase;
    const unsigned l = (unsigned)(unsigned long long)lbase;  // LDS offset = addr[31:0]
#pragma unroll
    for (int i = 0; i < 4; ++i) {
        const unsigned off = (unsigned)(tid * 16 + i * 2048);
        asm volatile("global_load_async_to_lds_b128 %0, %1, off"
                     :: "v"(l + off), "v"(g + (unsigned long long)off)
                     : "memory");
    }
}

// ---------------------------------------------------------------------------
// Kernel 1: f32 -> f16 convert
// ---------------------------------------------------------------------------
__global__ void cvt_f32_to_f16(const float* __restrict__ src,
                               _Float16* __restrict__ dst, int n) {
    int i = blockIdx.x * blockDim.x + threadIdx.x;
    if (i < n) dst[i] = (_Float16)src[i];
}

// ---------------------------------------------------------------------------
// Kernel 2: QKV projection.  out[b,h,s,d] = sum_k X[b,s,k] * W[h*128+d, k]
// M = B*S = 4096, N = 1024, K = 128.  One 16x16 tile per wave, 4 WMMAs.
// ---------------------------------------------------------------------------
__global__ __launch_bounds__(128)
void qkv_proj_kernel(const _Float16* __restrict__ xq,
                     const _Float16* __restrict__ xk,
                     const _Float16* __restrict__ xv,
                     const _Float16* __restrict__ wq,
                     const _Float16* __restrict__ wk,
                     const _Float16* __restrict__ wv,
                     _Float16* __restrict__ oq,
                     _Float16* __restrict__ ok,
                     _Float16* __restrict__ ov) {
    const _Float16* x;
    const _Float16* w;
    _Float16* dst;
    if (blockIdx.z == 0)      { x = xq; w = wq; dst = oq; }
    else if (blockIdx.z == 1) { x = xk; w = wk; dst = ok; }
    else                      { x = xv; w = wv; dst = ov; }

    const int tid  = threadIdx.x;
    const int wv32 = tid >> 5;
    const int lane = tid & 31;
    const int m15  = lane & 15;
    const int hi   = lane >> 4;

    const int tile = blockIdx.x * 4 + wv32;     // 16384 tiles total
    const int mt = tile >> 6;                   // 64 n-tiles
    const int nt = tile & 63;
    const int m0 = mt * 16, n0 = nt * 16;

    v8f acc = {};
#pragma unroll
    for (int c = 0; c < 4; ++c) {
        AFrag a, b;
#pragma unroll
        for (int v = 0; v < 8; ++v) {
            const int ka = 32 * c + ((v >> 2) << 4) + hi * 8 + ((v & 3) << 1);
            const int kb = 32 * c + ((v >> 2) << 4) + hi * 8 + ((v & 3) << 1);
            a.u[v] = *(const unsigned int*)(x + (size_t)(m0 + m15) * DIM + ka);
            b.u[v] = *(const unsigned int*)(w + (size_t)(n0 + m15) * DIM + kb);
        }
        acc = wmma16(a, b, acc);
    }

#pragma unroll
    for (int v = 0; v < 8; ++v) {
        const int m = m0 + hi * 8 + v;          // = b*S + s
        const int n = n0 + m15;                 // = h*128 + d
        const int bb = m >> 11, ss = m & (S_LEN - 1);
        const int hh = n >> 7,  dd = n & (DIM - 1);
        dst[(((size_t)(bb * HEADS + hh)) * S_LEN + ss) * DIM + dd] = (_Float16)acc[v];
    }
}

// ---------------------------------------------------------------------------
// Kernel 3: flash attention.  grid = (S/64, H, B), block = 128 (4 waves).
// Each wave owns 16 query rows; key blocks of 32, async double-buffered K/V.
// ---------------------------------------------------------------------------
__global__ __launch_bounds__(128)
void flash_attn_kernel(const _Float16* __restrict__ qh,
                       const _Float16* __restrict__ kh,
                       const _Float16* __restrict__ vh,
                       const int* __restrict__ mask,
                       _Float16* __restrict__ att16) {
    __shared__ __align__(16) _Float16 Kbuf[2][32 * DIM];   // [key][d]
    __shared__ __align__(16) _Float16 Vbuf[2][32 * DIM];   // [key][d] (linear)
    __shared__ __align__(16) _Float16 Pw[4][16 * 32];      // per-wave P staging

    const int tid  = threadIdx.x;
    const int wv32 = tid >> 5;
    const int lane = tid & 31;
    const int m15  = lane & 15;
    const int hi   = lane >> 4;

    const int b  = blockIdx.z;
    const int h  = blockIdx.y;
    const int q0 = blockIdx.x * 64 + wv32 * 16;

    const size_t headoff = ((size_t)(b * HEADS + h)) * S_LEN * DIM;
    const _Float16* qp = qh + headoff;
    const _Float16* kp = kh + headoff;
    const _Float16* vp = vh + headoff;

    // Resident Q fragments: 16 rows x 128 d = 4 A-fragments.
    AFrag aq[4];
#pragma unroll
    for (int c = 0; c < 4; ++c)
#pragma unroll
        for (int v = 0; v < 8; ++v) {
            const int k = 32 * c + ((v >> 2) << 4) + hi * 8 + ((v & 3) << 1);
            aq[c].u[v] = *(const unsigned int*)(qp + (size_t)(q0 + m15) * DIM + k);
        }

    v8f of[8];
#pragma unroll
    for (int f = 0; f < 8; ++f) of[f] = (v8f){};
    float rm[8], rl[8];
#pragma unroll
    for (int v = 0; v < 8; ++v) { rm[v] = -3.0e38f; rl[v] = 0.0f; }

    const float scl = 0.08838834764831845f;  // 1/sqrt(128)
    const int nkb = S_LEN / 32;

    // Prologue: async-stage block 0 into buffer 0.
    stage_tile_async(kp, &Kbuf[0][0], tid);
    stage_tile_async(vp, &Vbuf[0][0], tid);
    asm volatile("s_wait_asynccnt 0" ::: "memory");
    __syncthreads();

    for (int kb = 0; kb < nkb; ++kb) {
        const int cur = kb & 1;

        // Kick off the next block's async copies into the other buffer.
        if (kb + 1 < nkb) {
            stage_tile_async(kp + (size_t)(kb + 1) * 32 * DIM, &Kbuf[cur ^ 1][0], tid);
            stage_tile_async(vp + (size_t)(kb + 1) * 32 * DIM, &Vbuf[cur ^ 1][0], tid);
            __builtin_prefetch(mask + ((size_t)b * S_LEN + q0) * S_LEN + (kb + 1) * 32, 0, 1);
        }

        const _Float16* Kc = &Kbuf[cur][0];
        const _Float16* Vc = &Vbuf[cur][0];

        // ---- S = Q * K^T  (16 q x 32 keys, accumulate over d=128)
        v8f sc0 = {}, sc1 = {};
#pragma unroll
        for (int c = 0; c < 4; ++c) {
            AFrag bk0, bk1;
#pragma unroll
            for (int v = 0; v < 8; ++v) {
                const int d = 32 * c + ((v >> 2) << 4) + hi * 8 + ((v & 3) << 1);
                bk0.u[v] = *(const unsigned int*)(Kc + m15 * DIM + d);
                bk1.u[v] = *(const unsigned int*)(Kc + (16 + m15) * DIM + d);
            }
            sc0 = wmma16(aq[c], bk0, sc0);
            sc1 = wmma16(aq[c], bk1, sc1);
        }

        // ---- mask + online softmax (per-row stats, row = v + hi*8)
#pragma unroll
        for (int v = 0; v < 8; ++v) {
            const int q = q0 + hi * 8 + v;
            const int* mrow = mask + ((size_t)b * S_LEN + q) * S_LEN + kb * 32;
            float s0 = sc0[v], s1 = sc1[v];
            s0 = mrow[m15]      ? s0 * scl : -1.0e30f;
            s1 = mrow[16 + m15] ? s1 * scl : -1.0e30f;

            float rmax = fmaxf(s0, s1);
            rmax = fmaxf(rmax, __shfl_xor(rmax, 1, 32));
            rmax = fmaxf(rmax, __shfl_xor(rmax, 2, 32));
            rmax = fmaxf(rmax, __shfl_xor(rmax, 4, 32));
            rmax = fmaxf(rmax, __shfl_xor(rmax, 8, 32));

            const float mnew = fmaxf(rm[v], rmax);
            const float p0 = __expf(s0 - mnew);
            const float p1 = __expf(s1 - mnew);
            float rs = p0 + p1;
            rs += __shfl_xor(rs, 1, 32);
            rs += __shfl_xor(rs, 2, 32);
            rs += __shfl_xor(rs, 4, 32);
            rs += __shfl_xor(rs, 8, 32);

            const float cf = __expf(rm[v] - mnew);
            rm[v] = mnew;
            rl[v] = rl[v] * cf + rs;
#pragma unroll
            for (int f = 0; f < 8; ++f) of[f][v] *= cf;

            Pw[wv32][(v + hi * 8) * 32 + m15]      = (_Float16)p0;
            Pw[wv32][(v + hi * 8) * 32 + 16 + m15] = (_Float16)p1;
        }

        // wave-private LDS round trip: C-layout -> A-layout
        asm volatile("s_wait_dscnt 0" ::: "memory");

        AFrag pa;
#pragma unroll
        for (int v = 0; v < 8; ++v) {
            const int k = ((v >> 2) << 4) + hi * 8 + ((v & 3) << 1);
            pa.u[v] = *(const unsigned int*)(&Pw[wv32][m15 * 32 + k]);
        }

        // ---- O += P * V  (8 d-chunks of 16); V is linear [key][d] in LDS,
        // B-fragment needs (key, key+1) pairs at fixed d -> two u16 loads.
#pragma unroll
        for (int f = 0; f < 8; ++f) {
            AFrag bv;
#pragma unroll
            for (int v = 0; v < 8; ++v) {
                const int kk = ((v >> 2) << 4) + hi * 8 + ((v & 3) << 1);
                const int d  = f * 16 + m15;
                const unsigned lo  = *(const unsigned short*)(Vc + kk * DIM + d);
                const unsigned hi2 = *(const unsigned short*)(Vc + (kk + 1) * DIM + d);
                bv.u[v] = lo | (hi2 << 16);
            }
            of[f] = wmma16(pa, bv, of[f]);
        }

        // Next buffer must be resident on all waves before the next iteration.
        asm volatile("s_wait_asynccnt 0" ::: "memory");
        __syncthreads();
    }

    // ---- normalize + write (concat layout: att16[b, s, h*128 + d], f16)
#pragma unroll
    for (int v = 0; v < 8; ++v) {
        const float inv = 1.0f / rl[v];
        const int q = q0 + hi * 8 + v;
        _Float16* orow = att16 + ((size_t)b * S_LEN + q) * HDIM + h * DIM;
#pragma unroll
        for (int f = 0; f < 8; ++f)
            orow[f * 16 + m15] = (_Float16)(of[f][v] * inv);
    }
}

// ---------------------------------------------------------------------------
// Kernel 4: output projection.  out = att16 (4096x1024) @ Wu^T (128x1024) + bu
// ---------------------------------------------------------------------------
__global__ __launch_bounds__(128)
void out_proj_kernel(const _Float16* __restrict__ a16,
                     const _Float16* __restrict__ wu,
                     const float* __restrict__ bu,
                     float* __restrict__ out) {
    const int tid  = threadIdx.x;
    const int wv32 = tid >> 5;
    const int lane = tid & 31;
    const int m15  = lane & 15;
    const int hi   = lane >> 4;

    const int tile = blockIdx.x * 4 + wv32;     // 2048 tiles
    const int mt = tile >> 3;                   // 8 n-tiles
    const int nt = tile & 7;
    const int m0 = mt * 16, n0 = nt * 16;

    v8f acc = {};
    for (int c = 0; c < 32; ++c) {
        AFrag a, b;
#pragma unroll
        for (int v = 0; v < 8; ++v) {
            const int ka = 32 * c + ((v >> 2) << 4) + hi * 8 + ((v & 3) << 1);
            const int kb = 32 * c + ((v >> 2) << 4) + hi * 8 + ((v & 3) << 1);
            a.u[v] = *(const unsigned int*)(a16 + (size_t)(m0 + m15) * HDIM + ka);
            b.u[v] = *(const unsigned int*)(wu + (size_t)(n0 + m15) * HDIM + kb);
        }
        acc = wmma16(a, b, acc);
    }

#pragma unroll
    for (int v = 0; v < 8; ++v) {
        const int m = m0 + hi * 8 + v;
        const int n = n0 + m15;
        out[(size_t)m * DIM + n] = acc[v] + bu[n];
    }
}

// ---------------------------------------------------------------------------
// Host launcher
// ---------------------------------------------------------------------------
extern "C" void kernel_launch(void* const* d_in, const int* in_sizes, int n_in,
                              void* d_out, int out_size, void* d_ws, size_t ws_size,
                              hipStream_t stream) {
    const float* values = (const float*)d_in[0];
    const float* keys   = (const float*)d_in[1];
    const float* query  = (const float*)d_in[2];
    const int*   mask   = (const int*)d_in[3];
    const float* Wv     = (const float*)d_in[4];
    const float* Wk     = (const float*)d_in[5];
    const float* Wq     = (const float*)d_in[6];
    const float* Wu     = (const float*)d_in[7];
    const float* bu     = (const float*)d_in[8];
    float* out = (float*)d_out;

    const size_t N_X = (size_t)B_N * S_LEN * DIM;          // 524288
    const size_t N_W = (size_t)HDIM * DIM;                 // 131072
    const size_t N_H = (size_t)B_N * HEADS * S_LEN * DIM;  // 4194304
    const size_t N_A = (size_t)B_N * S_LEN * HDIM;         // 4194304

    _Float16* ws = (_Float16*)d_ws;
    size_t off = 0;
    _Float16* x16q = ws + off; off += N_X;
    _Float16* x16k = ws + off; off += N_X;
    _Float16* x16v = ws + off; off += N_X;
    _Float16* wq16 = ws + off; off += N_W;
    _Float16* wk16 = ws + off; off += N_W;
    _Float16* wv16 = ws + off; off += N_W;
    _Float16* wu16 = ws + off; off += N_W;
    _Float16* qh   = ws + off; off += N_H;
    _Float16* kh   = ws + off; off += N_H;
    _Float16* vh   = ws + off; off += N_H;
    _Float16* a16  = ws + off; off += N_A;

    // 1) convert activations + weights to f16
    cvt_f32_to_f16<<<(int)(N_X / 256), 256, 0, stream>>>(query,  x16q, (int)N_X);
    cvt_f32_to_f16<<<(int)(N_X / 256), 256, 0, stream>>>(keys,   x16k, (int)N_X);
    cvt_f32_to_f16<<<(int)(N_X / 256), 256, 0, stream>>>(values, x16v, (int)N_X);
    cvt_f32_to_f16<<<(int)(N_W / 256), 256, 0, stream>>>(Wq, wq16, (int)N_W);
    cvt_f32_to_f16<<<(int)(N_W / 256), 256, 0, stream>>>(Wk, wk16, (int)N_W);
    cvt_f32_to_f16<<<(int)(N_W / 256), 256, 0, stream>>>(Wv, wv16, (int)N_W);
    cvt_f32_to_f16<<<(int)(N_W / 256), 256, 0, stream>>>(Wu, wu16, (int)N_W);

    // 2) QKV projection: 16384 tiles / 4 waves per WG, z selects q/k/v
    qkv_proj_kernel<<<dim3(4096, 1, 3), 128, 0, stream>>>(
        x16q, x16k, x16v, wq16, wk16, wv16, qh, kh, vh);

    // 3) flash attention: (S/64, H, B) workgroups
    flash_attn_kernel<<<dim3(S_LEN / 64, HEADS, B_N), 128, 0, stream>>>(
        qh, kh, vh, mask, a16);

    // 4) output projection + bias: 2048 tiles / 4 waves per WG
    out_proj_kernel<<<512, 128, 0, stream>>>(a16, wu16, bu, out);
}
